// Attention_66340064854620
// MI455X (gfx1250) — matrix operand order repaired and checked by reading, not compiled, not run on hardware
//
#include <hip/hip_runtime.h>
#include <hip/hip_bf16.h>

#define BATCH 2
#define SEQL 2048
#define DMODEL 2048
#define NH 32
#define NKV 8
#define HD 64

typedef unsigned short u16;
typedef unsigned int u32;

typedef __attribute__((ext_vector_type(16))) __bf16 bf16x16;
typedef __attribute__((ext_vector_type(8))) float f32x8;
typedef __attribute__((ext_vector_type(4))) unsigned int u32x4;
typedef __attribute__((ext_vector_type(8))) int i32x8;
typedef __attribute__((ext_vector_type(4))) int i32x4;

#if __has_builtin(__builtin_amdgcn_tensor_load_to_lds) && __has_builtin(__builtin_amdgcn_s_wait_tensorcnt)
#define USE_TDM 1
#else
#define USE_TDM 0
#endif

union FragU { uint4 u[2]; bf16x16 v; };

__device__ __forceinline__ u16 f2bf(float f) {
  u32 u = __builtin_bit_cast(u32, f);
  u32 r = u + 0x7FFFu + ((u >> 16) & 1u);
  return (u16)(r >> 16);
}

__device__ __forceinline__ f32x8 zero8() {
  f32x8 z = {0.f,0.f,0.f,0.f,0.f,0.f,0.f,0.f};
  return z;
}

#if USE_TDM
// Issue one TDM 2D tile load: 64 rows x 64 bf16 elements, row stride
// `stride_elems` (in 2-byte units), into LDS at byte offset lds_off with the
// pad feature inserting +4 DWORDs every 32 DWORDs (=> 72-half LDS row stride).
// Descriptor bitfields per CDNA5 ISA.
__device__ __forceinline__ void tdm_load_tile64(u32 lds_off, const u16* gptr,
                                                u32 stride_elems) {
  unsigned long long ga = (unsigned long long)(size_t)(const void*)gptr;
  u32x4 g0;
  g0[0] = 1u;                                             // count=1 (valid), user mode
  g0[1] = lds_off;                                        // lds_addr (bytes)
  g0[2] = (u32)ga;                                        // global_addr[31:0]
  g0[3] = (u32)((ga >> 32) & 0x1FFFFFFu) | (2u << 30);    // global_addr[56:32], type=2
  i32x8 g1;
  g1[0] = (int)((1u << 16) |        // data_size = 1 (2 bytes)
                (1u << 20) |        // pad_enable
                (4u << 22) |        // pad_interval: 32 DWORDs
                (3u << 25));        // pad_amount:   4 DWORDs
  g1[1] = (int)(64u << 16);         // tensor_dim0[15:0]=64 (atomic_barrier_addr=0)
  g1[2] = (int)(64u << 16);         // tensor_dim0[31:16]=0, tensor_dim1[15:0]=64
  g1[3] = (int)(64u << 16);         // tensor_dim1[31:16]=0, tile_dim0=64
  g1[4] = (int)64u;                 // tile_dim1=64, tile_dim2=0
  g1[5] = (int)stride_elems;        // tensor_dim0_stride[31:0]
  g1[6] = 0;                        // stride hi, tensor_dim1_stride lo
  g1[7] = 0;
  i32x4 z4 = {0, 0, 0, 0};
#if defined(__clang_major__) && (__clang_major__ >= 23)
  i32x8 z8 = {0, 0, 0, 0, 0, 0, 0, 0};
  __builtin_amdgcn_tensor_load_to_lds(g0, g1, z4, z4, z8, 0);
#else
  __builtin_amdgcn_tensor_load_to_lds(g0, g1, z4, z4, 0);
#endif
}
#endif

// ---------------------------------------------------------------------------
// f32 -> bf16 converters
// ---------------------------------------------------------------------------
__global__ void cvt_bf16(const float* __restrict__ in, u16* __restrict__ out, size_t n) {
  size_t i = (size_t)blockIdx.x * blockDim.x + threadIdx.x;
  size_t stride = (size_t)gridDim.x * blockDim.x;
  for (; i < n; i += stride) out[i] = f2bf(in[i]);
}

// out[n][k] = bf16(in[k][n]) : transposed bf16 weights (N x K row-major)
__global__ void cvt_bf16_T(const float* __restrict__ in, u16* __restrict__ out,
                           int rows /*K*/, int cols /*N*/) {
  size_t n = (size_t)rows * cols;
  size_t i = (size_t)blockIdx.x * blockDim.x + threadIdx.x;
  size_t stride = (size_t)gridDim.x * blockDim.x;
  for (; i < n; i += stride) {
    int r = (int)(i / cols);
    int c = (int)(i % cols);
    out[(size_t)c * rows + r] = f2bf(in[i]);
  }
}

// ---------------------------------------------------------------------------
// WMMA GEMM: C[M,N] = A[M,K] (row-major bf16) @ BT[N,K]^T (bf16, K contiguous)
// 128x128 tile / 256-thread block, 8 waves of 32x64, K-step 32, software-
// pipelined global->reg loads overlapping WMMA.
// MODE 0: write Qbf  (B,NH,S,HD)    bf16
// MODE 1: write Kbf  (B,NKV,S,HD)   bf16
// MODE 2: write Vt   (B,NKV,HD,S)   bf16 (transposed store, packed b128)
// MODE 3: write f32  row-major MxN
// ---------------------------------------------------------------------------
template<int MODE>
__global__ __launch_bounds__(256) void gemm_bf16(const u16* __restrict__ A,
                                                 const u16* __restrict__ BT,
                                                 u16* __restrict__ outB,
                                                 float* __restrict__ outF,
                                                 int M, int N, int K) {
  __shared__ u16 As[128 * 40];
  __shared__ u16 Bs[128 * 40];
  const int tid  = threadIdx.x;
  const int w    = tid >> 5;
  const int lane = tid & 31;
  const int wm = w & 3, wn = w >> 2;
  const int m0 = blockIdx.y * 128;
  const int n0 = blockIdx.x * 128;
  const int nl = lane & 15, grp = lane >> 4, hi8 = grp * 8;

  f32x8 acc[2][4];
  for (int i = 0; i < 2; i++)
    for (int j = 0; j < 4; j++) acc[i][j] = zero8();

  const int lr = tid >> 1;        // 0..127
  const int lc = (tid & 1) * 16;  // halves
  const u16* gA = A  + (size_t)(m0 + lr) * K + lc;
  const u16* gB = BT + (size_t)(n0 + lr) * K + lc;
  u16* sA = &As[lr * 40 + lc];
  u16* sB = &Bs[lr * 40 + lc];

  uint4 a0 = *(const uint4*)(gA);
  uint4 a1 = *(const uint4*)(gA + 8);
  uint4 b0 = *(const uint4*)(gB);
  uint4 b1 = *(const uint4*)(gB + 8);

  for (int k0 = 0; k0 < K; k0 += 32) {
    __syncthreads();
    *(uint4*)sA = a0; *(uint4*)(sA + 8) = a1;
    *(uint4*)sB = b0; *(uint4*)(sB + 8) = b1;
    __syncthreads();

    if (k0 + 32 < K) {  // next-tile loads in flight during WMMA
      a0 = *(const uint4*)(gA + k0 + 32);
      a1 = *(const uint4*)(gA + k0 + 40);
      b0 = *(const uint4*)(gB + k0 + 32);
      b1 = *(const uint4*)(gB + k0 + 40);
      if (k0 + 64 < K) {
        __builtin_prefetch(gA + k0 + 64, 0, 1);
        __builtin_prefetch(gB + k0 + 64, 0, 1);
      }
    }

    FragU af[2];
    for (int sm = 0; sm < 2; sm++) {
      const u16* p = &As[(wm * 32 + sm * 16 + nl) * 40 + grp * 8];
      af[sm].u[0] = *(const uint4*)p;
      af[sm].u[1] = *(const uint4*)(p + 16);
    }
    for (int sn = 0; sn < 4; sn++) {
      FragU bf;
      const u16* p = &Bs[(wn * 64 + sn * 16 + nl) * 40 + grp * 8];
      bf.u[0] = *(const uint4*)p;
      bf.u[1] = *(const uint4*)(p + 16);
      for (int sm = 0; sm < 2; sm++) {
        acc[sm][sn] = __builtin_amdgcn_wmma_f32_16x16x32_bf16(
            false, af[sm].v, false, bf.v, (short)0, acc[sm][sn], false, false);
      }
    }
  }

  for (int sm = 0; sm < 2; sm++) {
    for (int sn = 0; sn < 4; sn++) {
      const int ng = n0 + wn * 64 + sn * 16 + nl;
      if (MODE == 0 || MODE == 1) {
        const int h = ng >> 6, d = ng & 63;
        const int heads = (MODE == 0) ? NH : NKV;
        for (int i = 0; i < 8; i++) {
          int mg = m0 + wm * 32 + sm * 16 + hi8 + i;
          int b = mg >> 11, t = mg & (SEQL - 1);
          size_t idx = ((size_t)(b * heads + h) * SEQL + t) * HD + d;
          outB[idx] = f2bf(acc[sm][sn][i]);
        }
      } else if (MODE == 2) {
        const int h = ng >> 6, d = ng & 63;
        int mg0 = m0 + wm * 32 + sm * 16 + hi8;
        int b = mg0 >> 11, t = mg0 & (SEQL - 1);
        u32 pk0 = (u32)f2bf(acc[sm][sn][0]) | ((u32)f2bf(acc[sm][sn][1]) << 16);
        u32 pk1 = (u32)f2bf(acc[sm][sn][2]) | ((u32)f2bf(acc[sm][sn][3]) << 16);
        u32 pk2 = (u32)f2bf(acc[sm][sn][4]) | ((u32)f2bf(acc[sm][sn][5]) << 16);
        u32 pk3 = (u32)f2bf(acc[sm][sn][6]) | ((u32)f2bf(acc[sm][sn][7]) << 16);
        size_t idx = ((size_t)(b * NKV + h) * HD + d) * SEQL + t;
        *(uint4*)(outB + idx) = make_uint4(pk0, pk1, pk2, pk3);
      } else {
        for (int i = 0; i < 8; i++) {
          int mg = m0 + wm * 32 + sm * 16 + hi8 + i;
          outF[(size_t)mg * N + ng] = acc[sm][sn][i];
        }
      }
    }
  }
}

// ---------------------------------------------------------------------------
// Flash attention (causal + graph_bias), GQA 32q/8kv heads, head_dim 64.
// 128 query rows per block; 8 waves each own a 16-row strip. K/V tiles are
// fetched by the Tensor Data Mover (double-buffered, overlapping WMMA),
// synchronized with s_wait_tensorcnt.
// ---------------------------------------------------------------------------
__global__ __launch_bounds__(256) void flash_attn(const u16* __restrict__ Q,
                                                  const u16* __restrict__ Kc,
                                                  const u16* __restrict__ Vt,
                                                  const float* __restrict__ gb,
                                                  u16* __restrict__ O) {
  __shared__ u16 Qs[128 * 72];
  __shared__ u16 Ks[2][64 * 72];
  __shared__ u16 Vs[2][64 * 72];
  __shared__ u16 Ps[128 * 72];

  const int tid = threadIdx.x;
  const int w = tid >> 5, lane = tid & 31;
  const int nl = lane & 15, grp = lane >> 4, hi8 = grp * 8;
  const int q0 = blockIdx.x * 128;
  const int h = blockIdx.y;
  const int b = blockIdx.z;
  const int kvh = h >> 2;  // N_REP = 4

  const int r0 = q0 + w * 16;
  const int nkb = blockIdx.x * 2 + 2;  // causal: key blocks up to the diagonal
  const u16* Kbase = Kc + (size_t)(b * NKV + kvh) * SEQL * HD;
  const u16* Vbase = Vt + (size_t)(b * NKV + kvh) * HD * SEQL;
  const float* gbase = gb + (size_t)b * SEQL * SEQL;

#if USE_TDM
  if (w == 0) {  // kick off K/V tiles for kb=0 while everyone loads Q
    tdm_load_tile64((u32)(size_t)(void*)&Ks[0][0], Kbase, HD);
    tdm_load_tile64((u32)(size_t)(void*)&Vs[0][0], Vbase, SEQL);
  }
#endif

  {  // cooperative Q tile load: 128 rows x 64 halves
    int r = tid >> 1, c = (tid & 1) * 32;
    const u16* src = Q + ((size_t)(b * NH + h) * SEQL + q0 + r) * HD + c;
    uint4 v0 = *(const uint4*)src;
    uint4 v1 = *(const uint4*)(src + 8);
    uint4 v2 = *(const uint4*)(src + 16);
    uint4 v3 = *(const uint4*)(src + 24);
    u16* d = &Qs[r * 72 + c];
    *(uint4*)d = v0; *(uint4*)(d + 8) = v1;
    *(uint4*)(d + 16) = v2; *(uint4*)(d + 24) = v3;
  }
  __syncthreads();

  FragU qf[2];
  {
    int r = w * 16 + nl;
    for (int ks = 0; ks < 2; ks++) {
      const u16* p = &Qs[r * 72 + ks * 32 + grp * 8];
      qf[ks].u[0] = *(const uint4*)p;
      qf[ks].u[1] = *(const uint4*)(p + 16);
    }
  }

  float mi[8], li[8];
  f32x8 oacc[4];
  for (int i = 0; i < 8; i++) { mi[i] = -1e30f; li[i] = 0.0f; }
  for (int dt = 0; dt < 4; dt++) oacc[dt] = zero8();

  for (int kb = 0; kb < nkb; kb++) {
    const int cur = kb & 1;
    __syncthreads();  // all waves done reading buffer (kb+1)&1 from iter kb-1
#if USE_TDM
    if (w == 0) {
      if (kb + 1 < nkb) {  // TDM for next block overlaps this block's WMMA
        tdm_load_tile64((u32)(size_t)(void*)&Ks[cur ^ 1][0],
                        Kbase + (size_t)(kb + 1) * 64 * HD, HD);
        tdm_load_tile64((u32)(size_t)(void*)&Vs[cur ^ 1][0],
                        Vbase + (size_t)(kb + 1) * 64, SEQL);
        __builtin_amdgcn_s_wait_tensorcnt(2);  // current block's tiles landed
      } else {
        __builtin_amdgcn_s_wait_tensorcnt(0);
      }
    }
#else
    {  // fallback: cooperative K (64x64 row=key) and V^T (64x64 row=d) loads
      int r = tid >> 2, c = (tid & 3) * 16;
      uint4 kv0 = *(const uint4*)(Kbase + (size_t)(kb * 64 + r) * HD + c);
      uint4 kv1 = *(const uint4*)(Kbase + (size_t)(kb * 64 + r) * HD + c + 8);
      uint4 vv0 = *(const uint4*)(Vbase + (size_t)r * SEQL + kb * 64 + c);
      uint4 vv1 = *(const uint4*)(Vbase + (size_t)r * SEQL + kb * 64 + c + 8);
      *(uint4*)&Ks[cur][r * 72 + c] = kv0; *(uint4*)&Ks[cur][r * 72 + c + 8] = kv1;
      *(uint4*)&Vs[cur][r * 72 + c] = vv0; *(uint4*)&Vs[cur][r * 72 + c + 8] = vv1;
    }
#endif
    __syncthreads();

    // S = Q @ K^T  (reduction over head_dim = 2 k-steps of 32)
    f32x8 sc[4];
    for (int nt = 0; nt < 4; nt++) {
      f32x8 c = zero8();
      for (int ks = 0; ks < 2; ks++) {
        FragU bf;
        const u16* p = &Ks[cur][(nt * 16 + nl) * 72 + ks * 32 + grp * 8];
        bf.u[0] = *(const uint4*)p;
        bf.u[1] = *(const uint4*)(p + 16);
        c = __builtin_amdgcn_wmma_f32_16x16x32_bf16(
            false, qf[ks].v, false, bf.v, (short)0, c, false, false);
      }
      sc[nt] = c;
    }

    // scale + causal mask + graph bias, row max
    float mrow[8];
    for (int i = 0; i < 8; i++) mrow[i] = -1e30f;
    for (int nt = 0; nt < 4; nt++) {
      int kg = kb * 64 + nt * 16 + nl;
      for (int i = 0; i < 8; i++) {
        int qg = r0 + hi8 + i;
        float v;
        if (kg <= qg) v = sc[nt][i] * 0.125f + gbase[(size_t)qg * SEQL + kg];
        else          v = -1e9f;
        sc[nt][i] = v;
        mrow[i] = fmaxf(mrow[i], v);
      }
    }
    for (int i = 0; i < 8; i++) {
      float m = mrow[i];
      m = fmaxf(m, __shfl_xor(m, 1, 32));
      m = fmaxf(m, __shfl_xor(m, 2, 32));
      m = fmaxf(m, __shfl_xor(m, 4, 32));
      m = fmaxf(m, __shfl_xor(m, 8, 32));
      mrow[i] = m;
    }
    float alpha[8];
    for (int i = 0; i < 8; i++) {
      float mn = fmaxf(mi[i], mrow[i]);
      alpha[i] = __expf(mi[i] - mn);
      mi[i] = mn;
    }
    float rs[8];
    for (int i = 0; i < 8; i++) rs[i] = 0.0f;
    for (int nt = 0; nt < 4; nt++)
      for (int i = 0; i < 8; i++) {
        float p = __expf(sc[nt][i] - mi[i]);
        sc[nt][i] = p;
        rs[i] += p;
      }
    for (int i = 0; i < 8; i++) {
      float s = rs[i];
      s += __shfl_xor(s, 1, 32);
      s += __shfl_xor(s, 2, 32);
      s += __shfl_xor(s, 4, 32);
      s += __shfl_xor(s, 8, 32);
      li[i] = li[i] * alpha[i] + s;
    }
    for (int dt = 0; dt < 4; dt++)
      for (int i = 0; i < 8; i++) oacc[dt][i] *= alpha[i];

    // re-stripe P: C-layout -> A-layout via LDS
    for (int nt = 0; nt < 4; nt++)
      for (int i = 0; i < 8; i++)
        Ps[(w * 16 + hi8 + i) * 72 + nt * 16 + nl] = f2bf(sc[nt][i]);
    __syncthreads();

    FragU pf[2];
    for (int ks = 0; ks < 2; ks++) {
      const u16* p = &Ps[(w * 16 + nl) * 72 + ks * 32 + grp * 8];
      pf[ks].u[0] = *(const uint4*)p;
      pf[ks].u[1] = *(const uint4*)(p + 16);
    }
    for (int dt = 0; dt < 4; dt++) {
      for (int ks = 0; ks < 2; ks++) {
        FragU bf;
        const u16* p = &Vs[cur][(dt * 16 + nl) * 72 + ks * 32 + grp * 8];
        bf.u[0] = *(const uint4*)p;
        bf.u[1] = *(const uint4*)(p + 16);
        oacc[dt] = __builtin_amdgcn_wmma_f32_16x16x32_bf16(
            false, pf[ks].v, false, bf.v, (short)0, oacc[dt], false, false);
      }
    }
  }

  // O /= l, store to (B, S, NH*HD) bf16
  for (int dt = 0; dt < 4; dt++) {
    int d = h * HD + dt * 16 + nl;
    for (int i = 0; i < 8; i++) {
      int qg = r0 + hi8 + i;
      float inv = 1.0f / li[i];
      O[((size_t)b * SEQL + qg) * DMODEL + d] = f2bf(oacc[dt][i] * inv);
    }
  }
}

// ---------------------------------------------------------------------------
extern "C" void kernel_launch(void* const* d_in, const int* in_sizes, int n_in,
                              void* d_out, int out_size, void* d_ws, size_t ws_size,
                              hipStream_t stream) {
  const float* x  = (const float*)d_in[0];
  // d_in[1] = mask: causal, computed analytically in-kernel
  const float* gb = (const float*)d_in[2];
  const float* wq = (const float*)d_in[3];
  const float* wk = (const float*)d_in[4];
  const float* wv = (const float*)d_in[5];
  const float* wo = (const float*)d_in[6];
  // d_in[7] = start_pos (0)
  float* out = (float*)d_out;

  char* ws = (char*)d_ws;
  size_t off = 0;
  auto alloc = [&](size_t bytes) { char* p = ws + off; off += bytes; return p; };
  u16* Xbf = (u16*)alloc((size_t)4096 * 2048 * 2);
  u16* WqT = (u16*)alloc((size_t)2048 * 2048 * 2);
  u16* WkT = (u16*)alloc((size_t)512 * 2048 * 2);
  u16* WvT = (u16*)alloc((size_t)512 * 2048 * 2);
  u16* WoT = (u16*)alloc((size_t)2048 * 2048 * 2);
  u16* Qbf = (u16*)alloc((size_t)BATCH * NH * SEQL * HD * 2);
  u16* Kbf = (u16*)alloc((size_t)BATCH * NKV * SEQL * HD * 2);
  u16* Vtb = (u16*)alloc((size_t)BATCH * NKV * HD * SEQL * 2);
  u16* Obf = (u16*)alloc((size_t)BATCH * SEQL * DMODEL * 2);

  cvt_bf16  <<<2048, 256, 0, stream>>>(x, Xbf, (size_t)4096 * 2048);
  cvt_bf16_T<<<2048, 256, 0, stream>>>(wq, WqT, 2048, 2048);
  cvt_bf16_T<<<512,  256, 0, stream>>>(wk, WkT, 2048, 512);
  cvt_bf16_T<<<512,  256, 0, stream>>>(wv, WvT, 2048, 512);
  cvt_bf16_T<<<2048, 256, 0, stream>>>(wo, WoT, 2048, 2048);

  dim3 blk(256);
  gemm_bf16<0><<<dim3(16, 32), blk, 0, stream>>>(Xbf, WqT, Qbf, nullptr, 4096, 2048, 2048);
  gemm_bf16<1><<<dim3(4,  32), blk, 0, stream>>>(Xbf, WkT, Kbf, nullptr, 4096, 512,  2048);
  gemm_bf16<2><<<dim3(4,  32), blk, 0, stream>>>(Xbf, WvT, Vtb, nullptr, 4096, 512,  2048);

  flash_attn<<<dim3(16, 32, 2), blk, 0, stream>>>(Qbf, Kbf, Vtb, gb, Obf);

  gemm_bf16<3><<<dim3(16, 32), blk, 0, stream>>>(Obf, WoT, nullptr, out, 4096, 2048, 2048);
}
